// Wavenet_38697655337504
// MI455X (gfx1250) — matrix operand order repaired and checked by reading, not compiled, not run on hardware
//
#include <hip/hip_runtime.h>
#include <hip/hip_bf16.h>

// ---------------- problem constants ----------------
#define TLEN 4096
#define RC   512
#define GC   512
#define SC   512
#define CIN  128
#define NLAY 30
#define FRONT 32
#define OUTC 2
#define SQRT_HALF 0.70710678118654752f

// K for gate GEMM: 512 (current H) + 512 (shifted H) + 128 (cond) = 1152 -> 36 k-tiles of 32
#define KT_GATE 36
#define KT_MS   16   // 512 / 32
#define TBLK    64   // timesteps per block (4 n-tiles of 16)

typedef __attribute__((ext_vector_type(16))) __bf16 bf16x16;
typedef __attribute__((ext_vector_type(8)))  __bf16 bf16x8;
typedef __attribute__((ext_vector_type(8)))  float  f32x8;

static __device__ __forceinline__ f32x8 wmma_bf16(bf16x16 a, bf16x16 b, f32x8 c) {
    // (neg_a, A, neg_b, B, c_mod, C, reuse_a, reuse_b)
    return __builtin_amdgcn_wmma_f32_16x16x32_bf16(false, a, false, b, (short)0, c, false, false);
}

// =====================================================================
// Weight packing: fp32 -> bf16, swizzled into this code's 16x32 A-matrix
// per-lane layout so GEMM A-loads are one contiguous 32B read per lane.
// A element for (lane L, vec elem j):  m = m0 + (L&15),
//   k = k0 + 16*(L>>4) + j
// =====================================================================
__global__ void pack_gate_kernel(const float* __restrict__ filt_w,
                                 const float* __restrict__ gate_w,
                                 const float* __restrict__ cf_w,
                                 const float* __restrict__ cg_w,
                                 __bf16* __restrict__ dst)
{
    size_t idx = (size_t)blockIdx.x * 256 + threadIdx.x;
    const size_t total = (size_t)NLAY * 2 * 32 * KT_GATE * 512;
    if (idx >= total) return;
    int j    = (int)(idx & 15);
    int lane = (int)((idx >> 4) & 31);
    size_t tile = idx >> 9;
    int kt    = (int)(tile % KT_GATE);
    int mt    = (int)((tile / KT_GATE) % 32);
    int sel   = (int)((tile / (KT_GATE * 32)) % 2);
    int layer = (int)(tile / (KT_GATE * 32 * 2));
    int m = mt * 16 + (lane & 15);
    int k = kt * 32 + 16 * (lane >> 4) + j;
    float v;
    if (k < 1024) {
        const float* w = sel ? gate_w : filt_w;     // (NL, GC, RC, 2)
        int kc = k & 511;
        int ks = (k < 512) ? 1 : 0;                 // col<512: current tap, else shifted tap
        v = w[(((size_t)layer * GC + m) * RC + kc) * 2 + ks];
    } else {
        const float* w = sel ? cg_w : cf_w;         // (NL, GC, CIN)
        v = w[((size_t)layer * GC + m) * CIN + (k - 1024)];
    }
    dst[idx] = (__bf16)v;
}

// mats: 0..29 res_w, 30..59 skip_w, 60 fin1_w  (all 512x512 row-major [out][in])
__global__ void pack_ms_kernel(const float* __restrict__ res_w,
                               const float* __restrict__ skip_w,
                               const float* __restrict__ fin1_w,
                               __bf16* __restrict__ dst)
{
    size_t idx = (size_t)blockIdx.x * 256 + threadIdx.x;
    const size_t total = (size_t)61 * 32 * KT_MS * 512;
    if (idx >= total) return;
    int j    = (int)(idx & 15);
    int lane = (int)((idx >> 4) & 31);
    size_t tile = idx >> 9;
    int kt  = (int)(tile % KT_MS);
    int mt  = (int)((tile / KT_MS) % 32);
    int mat = (int)(tile / (KT_MS * 32));
    int m = mt * 16 + (lane & 15);
    int k = kt * 32 + 16 * (lane >> 4) + j;
    const float* src;
    size_t base;
    if (mat < 30)      { src = res_w;  base = (size_t)mat * 512 * 512; }
    else if (mat < 60) { src = skip_w; base = (size_t)(mat - 30) * 512 * 512; }
    else               { src = fin1_w; base = 0; }
    dst[idx] = (__bf16)src[base + (size_t)m * 512 + k];
}

// =====================================================================
// Upsample (two transposed convs, scale 16 each, kernel 3x32, leaky 0.4)
// =====================================================================
static __device__ float up_conv(const float* __restrict__ I, int Win,
                                const float* __restrict__ K96, int h, int w)
{
    float acc = 0.f;
    int b0 = (23 - w) % 16;
    if (b0 < 0) b0 += 16;
    #pragma unroll
    for (int a = 0; a < 3; ++a) {
        int hh = h + a - 1;
        if (hh < 0 || hh >= 128) continue;
        for (int b = b0; b < 32; b += 16) {
            int q = w + b - 23;
            if (q < 0) continue;
            int xi = q >> 4;
            if (xi >= Win) continue;
            acc += K96[(2 - a) * 32 + (31 - b)] * I[hh * Win + xi];
        }
    }
    return acc;
}

__global__ void upsample1_kernel(const float* __restrict__ c,
                                 const float* __restrict__ upw,
                                 const float* __restrict__ upb,
                                 float* __restrict__ S1)
{
    int idx = blockIdx.x * 256 + threadIdx.x;
    if (idx >= 128 * 256) return;
    int h = idx / 256, w = idx % 256;
    float v = up_conv(c, 16, upw, h, w) + upb[0];
    S1[idx] = v > 0.f ? v : 0.4f * v;
}

// writes conditioning transposed (time-major, bf16): Cup[t][cin]
__global__ void upsample2_kernel(const float* __restrict__ S1,
                                 const float* __restrict__ upw,
                                 const float* __restrict__ upb,
                                 __bf16* __restrict__ Cup)
{
    int idx = blockIdx.x * 256 + threadIdx.x;
    if (idx >= 128 * TLEN) return;
    int h = idx / TLEN, w = idx % TLEN;
    float v = up_conv(S1, 256, upw + 96, h, w) + upb[1];
    v = v > 0.f ? v : 0.4f * v;
    Cup[(size_t)w * CIN + h] = (__bf16)v;
}

// =====================================================================
// Front causal conv (kernel 32), relu; time-major outputs H fp32 + bf16
// =====================================================================
__global__ void front_kernel(const float* __restrict__ x,
                             const float* __restrict__ fw,
                             const float* __restrict__ fb,
                             float* __restrict__ H,
                             __bf16* __restrict__ Hb)
{
    size_t idx = (size_t)blockIdx.x * 256 + threadIdx.x;
    if (idx >= (size_t)TLEN * RC) return;
    int t = (int)(idx >> 9), m = (int)(idx & 511);
    float acc = fb[m];
    const float* w = fw + m * FRONT;
    #pragma unroll 8
    for (int k = 0; k < FRONT; ++k) {
        int xt = t - (FRONT - 1) + k;
        if (xt >= 0) acc += w[k] * x[xt];
    }
    acc = acc > 0.f ? acc : 0.f;
    H[idx]  = acc;
    Hb[idx] = (__bf16)acc;
}

__global__ void zero_kernel(float* __restrict__ p, size_t n)
{
    size_t i = (size_t)blockIdx.x * 256 + threadIdx.x;
    if (i < n) p[i] = 0.f;
}

// =====================================================================
// Gate GEMM: F/G = W(1152-wide packed) x [H ; H(t-d) ; Cup], then
// o = tanh(F)*sigmoid(G) stored bf16 time-major. Each wave: 16ch x 64t.
// =====================================================================
__global__ __launch_bounds__(256)
void gate_kernel(const __bf16* __restrict__ Wf, const __bf16* __restrict__ Wg,
                 const __bf16* __restrict__ Hb, const __bf16* __restrict__ Cup,
                 const float* __restrict__ fb, const float* __restrict__ gb,
                 __bf16* __restrict__ O, int dil)
{
    const int lane = threadIdx.x & 31;
    const int wave = threadIdx.x >> 5;
    const int lm = lane & 15, half = lane >> 4;
    const int mt = blockIdx.y * 8 + wave;
    const int m0 = mt * 16;
    const int t0 = blockIdx.x * TBLK;

    int trow[4];
    #pragma unroll
    for (int nt = 0; nt < 4; ++nt) trow[nt] = t0 + nt * 16 + lm;

    f32x8 accF[4] = {}, accG[4] = {};
    const __bf16* af = Wf + ((size_t)mt * KT_GATE) * 512 + lane * 16;
    const __bf16* ag = Wg + ((size_t)mt * KT_GATE) * 512 + lane * 16;

    // region 1: K=0..511, current H
    #pragma unroll 2
    for (int kt = 0; kt < 16; ++kt) {
        int col = kt * 32 + 16 * half;
        bf16x16 a0 = *(const bf16x16*)af; af += 512;
        bf16x16 a1 = *(const bf16x16*)ag; ag += 512;
        #pragma unroll
        for (int nt = 0; nt < 4; ++nt) {
            bf16x16 b = *(const bf16x16*)(Hb + (size_t)trow[nt] * 512 + col);
            accF[nt] = wmma_bf16(a0, b, accF[nt]);
            accG[nt] = wmma_bf16(a1, b, accG[nt]);
        }
    }
    // region 2: K=512..1023, dilated-shift H (zero below t=0)
    #pragma unroll 2
    for (int kt = 0; kt < 16; ++kt) {
        int col = kt * 32 + 16 * half;
        bf16x16 a0 = *(const bf16x16*)af; af += 512;
        bf16x16 a1 = *(const bf16x16*)ag; ag += 512;
        #pragma unroll
        for (int nt = 0; nt < 4; ++nt) {
            int s = trow[nt] - dil;
            bf16x16 b = {};
            if (s >= 0) b = *(const bf16x16*)(Hb + (size_t)s * 512 + col);
            accF[nt] = wmma_bf16(a0, b, accF[nt]);
            accG[nt] = wmma_bf16(a1, b, accG[nt]);
        }
    }
    // region 3: K=1024..1151, conditioning
    #pragma unroll
    for (int kt = 0; kt < 4; ++kt) {
        int col = kt * 32 + 16 * half;
        bf16x16 a0 = *(const bf16x16*)af; af += 512;
        bf16x16 a1 = *(const bf16x16*)ag; ag += 512;
        #pragma unroll
        for (int nt = 0; nt < 4; ++nt) {
            bf16x16 b = *(const bf16x16*)(Cup + (size_t)trow[nt] * CIN + col);
            accF[nt] = wmma_bf16(a0, b, accF[nt]);
            accG[nt] = wmma_bf16(a1, b, accG[nt]);
        }
    }

    // epilogue: bias + tanh*sigmoid, store bf16 time-major (contiguous 16B/lane)
    #pragma unroll
    for (int nt = 0; nt < 4; ++nt) {
        bf16x8 ov;
        #pragma unroll
        for (int v = 0; v < 8; ++v) {
            int m = m0 + 8 * half + v;
            float f = accF[nt][v] + fb[m];
            float g = accG[nt][v] + gb[m];
            float o = tanhf(f) * (1.f / (1.f + expf(-g)));
            ov[v] = (__bf16)o;
        }
        *(bf16x8*)(O + (size_t)trow[nt] * 512 + m0 + 8 * half) = ov;
    }
}

// =====================================================================
// Res/skip GEMMs: H = (H + Wr*o + rb)*sqrt(0.5); S += Ws*o + sb
// =====================================================================
__global__ __launch_bounds__(256)
void resskip_kernel(const __bf16* __restrict__ Wr, const __bf16* __restrict__ Wsk,
                    const __bf16* __restrict__ O,
                    const float* __restrict__ rb, const float* __restrict__ sb,
                    float* __restrict__ H, __bf16* __restrict__ Hb,
                    float* __restrict__ S)
{
    const int lane = threadIdx.x & 31;
    const int wave = threadIdx.x >> 5;
    const int lm = lane & 15, half = lane >> 4;
    const int mt = blockIdx.y * 8 + wave;
    const int m0 = mt * 16;
    const int t0 = blockIdx.x * TBLK;

    int trow[4];
    #pragma unroll
    for (int nt = 0; nt < 4; ++nt) trow[nt] = t0 + nt * 16 + lm;

    f32x8 aR[4] = {}, aS[4] = {};
    const __bf16* ar = Wr  + ((size_t)mt * KT_MS) * 512 + lane * 16;
    const __bf16* as = Wsk + ((size_t)mt * KT_MS) * 512 + lane * 16;

    #pragma unroll 2
    for (int kt = 0; kt < KT_MS; ++kt) {
        int col = kt * 32 + 16 * half;
        bf16x16 a0 = *(const bf16x16*)ar; ar += 512;
        bf16x16 a1 = *(const bf16x16*)as; as += 512;
        #pragma unroll
        for (int nt = 0; nt < 4; ++nt) {
            bf16x16 b = *(const bf16x16*)(O + (size_t)trow[nt] * 512 + col);
            aR[nt] = wmma_bf16(a0, b, aR[nt]);
            aS[nt] = wmma_bf16(a1, b, aS[nt]);
        }
    }

    #pragma unroll
    for (int nt = 0; nt < 4; ++nt) {
        size_t base = (size_t)trow[nt] * 512 + m0 + 8 * half;
        #pragma unroll
        for (int v = 0; v < 8; ++v) {
            int m = m0 + 8 * half + v;
            float h = (H[base + v] + aR[nt][v] + rb[m]) * SQRT_HALF;
            H[base + v]  = h;
            Hb[base + v] = (__bf16)h;
            S[base + v] += aS[nt][v] + sb[m];
        }
    }
}

// relu(S) -> bf16 (feeds final1 GEMM)
__global__ void relu_bf16_kernel(const float* __restrict__ S, __bf16* __restrict__ Ob)
{
    size_t i = (size_t)blockIdx.x * 256 + threadIdx.x;
    if (i >= (size_t)TLEN * SC) return;
    float v = S[i];
    Ob[i] = (__bf16)(v > 0.f ? v : 0.f);
}

// Z = relu(fin1_w * relu(S) + fin1_b), stored bf16 time-major
__global__ __launch_bounds__(256)
void final1_kernel(const __bf16* __restrict__ W1, const __bf16* __restrict__ Bsrc,
                   const float* __restrict__ b1, __bf16* __restrict__ Z)
{
    const int lane = threadIdx.x & 31;
    const int wave = threadIdx.x >> 5;
    const int lm = lane & 15, half = lane >> 4;
    const int mt = blockIdx.y * 8 + wave;
    const int m0 = mt * 16;
    const int t0 = blockIdx.x * TBLK;

    int trow[4];
    #pragma unroll
    for (int nt = 0; nt < 4; ++nt) trow[nt] = t0 + nt * 16 + lm;

    f32x8 acc[4] = {};
    const __bf16* a = W1 + ((size_t)mt * KT_MS) * 512 + lane * 16;
    #pragma unroll 2
    for (int kt = 0; kt < KT_MS; ++kt) {
        int col = kt * 32 + 16 * half;
        bf16x16 aa = *(const bf16x16*)a; a += 512;
        #pragma unroll
        for (int nt = 0; nt < 4; ++nt) {
            bf16x16 b = *(const bf16x16*)(Bsrc + (size_t)trow[nt] * 512 + col);
            acc[nt] = wmma_bf16(aa, b, acc[nt]);
        }
    }
    #pragma unroll
    for (int nt = 0; nt < 4; ++nt) {
        bf16x8 ov;
        #pragma unroll
        for (int v = 0; v < 8; ++v) {
            int m = m0 + 8 * half + v;
            float z = acc[nt][v] + b1[m];
            ov[v] = (__bf16)(z > 0.f ? z : 0.f);
        }
        *(bf16x8*)(Z + (size_t)trow[nt] * 512 + m0 + 8 * half) = ov;
    }
}

// out[o][t] = fin2_w[o] . Z[t] + fin2_b[o]   (M=2 -> scalar kernel)
__global__ void final2_kernel(const __bf16* __restrict__ Z,
                              const float* __restrict__ w2,
                              const float* __restrict__ b2,
                              float* __restrict__ out)
{
    int idx = blockIdx.x * 256 + threadIdx.x;
    if (idx >= OUTC * TLEN) return;
    int o = idx / TLEN, t = idx % TLEN;
    float acc = b2[o];
    const __bf16* z = Z + (size_t)t * 512;
    const float* w = w2 + o * 512;
    #pragma unroll 8
    for (int c = 0; c < 512; ++c) acc += w[c] * (float)z[c];
    out[idx] = acc;
}

// =====================================================================
extern "C" void kernel_launch(void* const* d_in, const int* in_sizes, int n_in,
                              void* d_out, int out_size, void* d_ws, size_t ws_size,
                              hipStream_t stream)
{
    const float* x       = (const float*)d_in[0];
    const float* c       = (const float*)d_in[1];
    const float* front_w = (const float*)d_in[2];
    const float* front_b = (const float*)d_in[3];
    const float* filt_w  = (const float*)d_in[4];
    const float* filt_b  = (const float*)d_in[5];
    const float* gate_w  = (const float*)d_in[6];
    const float* gate_b  = (const float*)d_in[7];
    const float* cf_w    = (const float*)d_in[8];
    const float* cg_w    = (const float*)d_in[10];
    const float* res_w   = (const float*)d_in[12];
    const float* res_b   = (const float*)d_in[13];
    const float* skip_w  = (const float*)d_in[14];
    const float* skip_b  = (const float*)d_in[15];
    const float* fin1_w  = (const float*)d_in[16];
    const float* fin1_b  = (const float*)d_in[17];
    const float* fin2_w  = (const float*)d_in[18];
    const float* fin2_b  = (const float*)d_in[19];
    const float* up_w    = (const float*)d_in[20];
    const float* up_b    = (const float*)d_in[21];

    char* ws = (char*)d_ws;
    size_t off = 0;
    float*  Hf  = (float*)(ws + off);  off += (size_t)TLEN * RC * 4;      // residual fp32
    float*  S   = (float*)(ws + off);  off += (size_t)TLEN * SC * 4;      // skip accum fp32
    __bf16* Hb  = (__bf16*)(ws + off); off += (size_t)TLEN * RC * 2;      // residual bf16
    __bf16* Ob  = (__bf16*)(ws + off); off += (size_t)TLEN * GC * 2;      // gate output bf16
    __bf16* Cup = (__bf16*)(ws + off); off += (size_t)TLEN * CIN * 2;     // conditioning bf16
    float*  S1  = (float*)(ws + off);  off += (size_t)128 * 256 * 4;      // upsample stage 1
    __bf16* PackG = (__bf16*)(ws + off); off += (size_t)NLAY * 2 * 32 * KT_GATE * 512 * 2;
    __bf16* PackM = (__bf16*)(ws + off); off += (size_t)61 * 32 * KT_MS * 512 * 2;
    (void)ws_size; (void)in_sizes; (void)n_in; (void)out_size;

    // one-shot weight packing (runs every call; deterministic)
    {
        size_t n = (size_t)NLAY * 2 * 32 * KT_GATE * 512;
        pack_gate_kernel<<<(unsigned)((n + 255) / 256), 256, 0, stream>>>(filt_w, gate_w, cf_w, cg_w, PackG);
    }
    {
        size_t n = (size_t)61 * 32 * KT_MS * 512;
        pack_ms_kernel<<<(unsigned)((n + 255) / 256), 256, 0, stream>>>(res_w, skip_w, fin1_w, PackM);
    }

    upsample1_kernel<<<(128 * 256 + 255) / 256, 256, 0, stream>>>(c, up_w, up_b, S1);
    upsample2_kernel<<<(128 * TLEN + 255) / 256, 256, 0, stream>>>(S1, up_w, up_b, Cup);
    front_kernel<<<(unsigned)(((size_t)TLEN * RC + 255) / 256), 256, 0, stream>>>(x, front_w, front_b, Hf, Hb);
    zero_kernel<<<(unsigned)(((size_t)TLEN * SC + 255) / 256), 256, 0, stream>>>(S, (size_t)TLEN * SC);

    dim3 ggrid(TLEN / TBLK, GC / 128);  // (64, 4): 64 timesteps x 128 channels per block
    for (int i = 0; i < NLAY; ++i) {
        int dil = 1 << (i % 10);
        const __bf16* Wf  = PackG + (size_t)(i * 2 + 0) * 32 * KT_GATE * 512;
        const __bf16* Wg  = PackG + (size_t)(i * 2 + 1) * 32 * KT_GATE * 512;
        gate_kernel<<<ggrid, 256, 0, stream>>>(Wf, Wg, Hb, Cup,
                                               filt_b + (size_t)i * GC, gate_b + (size_t)i * GC,
                                               Ob, dil);
        const __bf16* Wr  = PackM + (size_t)i * 32 * KT_MS * 512;
        const __bf16* Wsk = PackM + (size_t)(30 + i) * 32 * KT_MS * 512;
        resskip_kernel<<<ggrid, 256, 0, stream>>>(Wr, Wsk, Ob,
                                                  res_b + (size_t)i * RC, skip_b + (size_t)i * SC,
                                                  Hf, Hb, S);
    }

    relu_bf16_kernel<<<(unsigned)(((size_t)TLEN * SC + 255) / 256), 256, 0, stream>>>(S, Ob);
    final1_kernel<<<ggrid, 256, 0, stream>>>(PackM + (size_t)60 * 32 * KT_MS * 512, Ob, fin1_b, Hb);
    final2_kernel<<<(OUTC * TLEN + 255) / 256, 256, 0, stream>>>(Hb, fin2_w, fin2_b, (float*)d_out);
}
